// Causal_attention_70308614635806
// MI455X (gfx1250) — matrix operand order repaired and checked by reading, not compiled, not run on hardware
//
#include <hip/hip_runtime.h>
#include <hip/hip_bf16.h>

// ---------------------------------------------------------------------------
// Causal attention, B=4, S=4096, D=512, fp32 in/out.
// Strategy: f16 WMMA (v_wmma_f32_16x16x32_f16) with f32 accumulation.
//   Kernel 1: QKV projection GEMM -> f16 Q (pre-scaled by 1/sqrt(D)), f16 K
//             (row-major), f16 V stored transposed [B][D][S].
//   Kernel 2: flash-attention, 4 waves/block, wave = 16 query rows x 512 dims,
//             double-buffered K/V^T tiles staged with async global->LDS copies
//             (ASYNCcnt-tracked, gfx1250).
// ---------------------------------------------------------------------------

typedef __attribute__((ext_vector_type(16))) _Float16 v16h;
typedef __attribute__((ext_vector_type(8)))  float    v8f;
typedef int v4i_vs __attribute__((vector_size(16)));   // int4, matches builtin

typedef __attribute__((address_space(1))) v4i_vs* glb_v4i_ptr;
typedef __attribute__((address_space(3))) v4i_vs* lds_v4i_ptr;

#define BATCH 4
#define SEQ   4096
#define DIM   512
#define MTOT  (BATCH * SEQ)          // 16384 rows
#define SOFTMAX_SCALE 0.04419417382415922f  // 1/sqrt(512)

// LDS strides (in halves) chosen for bank-conflict-free 16-lane access:
// 520*2B = 1040B = 260 words, 260 % 64 = 4  -> lanes hit distinct banks
//  40*2B =   80B =  20 words,  20 % 64 = 20 -> lanes hit distinct banks
#define QK_STRIDE 520
#define V_STRIDE  40
#define P_STRIDE  40
#define GEMM_STRIDE 40

// ---------------------------------------------------------------------------
// Async global -> LDS copy (GLOBAL_LOAD_ASYNC_TO_LDS_B128, ASYNCcnt).
// AS(3) pointers are 32-bit LDS offsets == low 32 bits of the generic address
// (aperture rule); AS(1) is numerically identical to the generic address.
// ---------------------------------------------------------------------------
__device__ __forceinline__ void async_copy_b128(const _Float16* gsrc,
                                                _Float16* ldst) {
  __builtin_amdgcn_global_load_async_to_lds_b128(
      (glb_v4i_ptr)(unsigned long long)(uintptr_t)gsrc,
      (lds_v4i_ptr)(unsigned int)(uintptr_t)ldst,
      0, 0);
}

__device__ __forceinline__ void async_wait0() {
  __builtin_amdgcn_s_wait_asynccnt(0);
}

// ---------------------------------------------------------------------------
// Fragment loaders (ISA 7.12.2 layouts, wave32).
// A (16x32 f16): lane m = lane&15, half = lane>>4.
//   VGPR r holds K pair: k0 = (r>>2)*16 + (r&3)*2 + half*8.
// B (32x16 f16): lane n = lane&15, half = lane>>4.
//   VGPR r holds K pair: k0 = (r>>2)*16 + (r&3)*2 + half*8 (same as A).
// ---------------------------------------------------------------------------
__device__ __forceinline__ v16h load_frag_a_lds(const _Float16* p, int stride,
                                                int mbase, int koff, int lane) {
  const int m  = mbase + (lane & 15);
  const int hi = lane >> 4;
  v16h f;
  unsigned int* fr = reinterpret_cast<unsigned int*>(&f);
  const _Float16* row = p + m * stride + koff + (hi << 3);
#pragma unroll
  for (int r = 0; r < 8; ++r) {
    const int k0 = ((r >> 2) << 4) + ((r & 3) << 1);
    fr[r] = *reinterpret_cast<const unsigned int*>(row + k0);
  }
  return f;
}

__device__ __forceinline__ v16h load_frag_a_glb(const _Float16* p, int stride,
                                                int koff, int lane) {
  const int m  = lane & 15;
  const int hi = lane >> 4;
  v16h f;
  unsigned int* fr = reinterpret_cast<unsigned int*>(&f);
  const _Float16* row = p + (size_t)m * stride + koff + (hi << 3);
#pragma unroll
  for (int r = 0; r < 8; ++r) {
    const int k0 = ((r >> 2) << 4) + ((r & 3) << 1);
    fr[r] = *reinterpret_cast<const unsigned int*>(row + k0);
  }
  return f;
}

__device__ __forceinline__ v16h load_frag_b_lds(const _Float16* p, int stride,
                                                int nbase, int koff, int lane) {
  const int n  = nbase + (lane & 15);
  const int hi = lane >> 4;
  v16h f;
  unsigned int* fr = reinterpret_cast<unsigned int*>(&f);
  const _Float16* row = p + n * stride + koff + (hi << 3);
#pragma unroll
  for (int r = 0; r < 8; ++r) {
    const int k0 = ((r >> 2) << 4) + ((r & 3) << 1);
    fr[r] = *reinterpret_cast<const unsigned int*>(row + k0);
  }
  return f;
}

#define WMMA_F16(A, B, C) \
  __builtin_amdgcn_wmma_f32_16x16x32_f16(false, (A), false, (B), (short)0, (C), false, false)

// ---------------------------------------------------------------------------
// Kernel 1: QKV projection.  y = x @ W^T, M=16384, N=512, K=512.
// grid = (2048, 3): blockIdx.x = mblk*8 + nblk, blockIdx.y = 0:q 1:k 2:v.
// Block = 128 threads (4 waves); each wave computes a 16x64 tile.
// ---------------------------------------------------------------------------
__global__ __launch_bounds__(128) void qkv_proj_kernel(
    const float* __restrict__ x,
    const float* __restrict__ wq, const float* __restrict__ wk,
    const float* __restrict__ wv,
    _Float16* __restrict__ qo, _Float16* __restrict__ ko,
    _Float16* __restrict__ vt) {
  __shared__ _Float16 Xs[64 * GEMM_STRIDE];
  __shared__ _Float16 Ws[64 * GEMM_STRIDE];

  const int tid  = threadIdx.x;
  const int wave = tid >> 5;
  const int lane = tid & 31;
  const int mb   = (blockIdx.x >> 3) * 64;
  const int nb   = (blockIdx.x & 7) * 64;
  const int mode = blockIdx.y;
  const float* __restrict__ w = (mode == 0) ? wq : (mode == 1) ? wk : wv;

  v8f acc[4] = {};

  // Cooperative staging: thread t loads 16 f32 -> 16 f16 per tile per chunk.
  const int srow = tid >> 1;
  const int scol = (tid & 1) << 4;

  for (int kc = 0; kc < DIM; kc += 32) {
    {
      const float4* src = reinterpret_cast<const float4*>(
          x + (size_t)(mb + srow) * DIM + kc + scol);
      _Float16* dst = Xs + srow * GEMM_STRIDE + scol;
#pragma unroll
      for (int i = 0; i < 4; ++i) {
        float4 v4 = src[i];
        dst[i * 4 + 0] = (_Float16)v4.x;
        dst[i * 4 + 1] = (_Float16)v4.y;
        dst[i * 4 + 2] = (_Float16)v4.z;
        dst[i * 4 + 3] = (_Float16)v4.w;
      }
    }
    {
      const float4* src = reinterpret_cast<const float4*>(
          w + (size_t)(nb + srow) * DIM + kc + scol);
      _Float16* dst = Ws + srow * GEMM_STRIDE + scol;
#pragma unroll
      for (int i = 0; i < 4; ++i) {
        float4 v4 = src[i];
        dst[i * 4 + 0] = (_Float16)v4.x;
        dst[i * 4 + 1] = (_Float16)v4.y;
        dst[i * 4 + 2] = (_Float16)v4.z;
        dst[i * 4 + 3] = (_Float16)v4.w;
      }
    }
    __syncthreads();

    v16h a = load_frag_a_lds(Xs, GEMM_STRIDE, wave * 16, 0, lane);
#pragma unroll
    for (int nt = 0; nt < 4; ++nt) {
      v16h b = load_frag_b_lds(Ws, GEMM_STRIDE, nt * 16, 0, lane);
      acc[nt] = WMMA_F16(a, b, acc[nt]);
    }
    __syncthreads();
  }

  const int hi = lane >> 4;
  const int ln = lane & 15;
  if (mode < 2) {
    _Float16* __restrict__ out = (mode == 0) ? qo : ko;
    const float sc = (mode == 0) ? SOFTMAX_SCALE : 1.0f;
#pragma unroll
    for (int nt = 0; nt < 4; ++nt) {
#pragma unroll
      for (int r = 0; r < 8; ++r) {
        const int m = mb + wave * 16 + r + 8 * hi;
        const int n = nb + nt * 16 + ln;
        out[(size_t)m * DIM + n] = (_Float16)(acc[nt][r] * sc);
      }
    }
  } else {
    // V transposed: vt[b][n][s]
#pragma unroll
    for (int nt = 0; nt < 4; ++nt) {
#pragma unroll
      for (int r = 0; r < 8; ++r) {
        const int m = mb + wave * 16 + r + 8 * hi;
        const int bb = m >> 12;          // / 4096
        const int ss = m & 4095;
        const int n = nb + nt * 16 + ln;
        vt[((size_t)bb * DIM + n) * SEQ + ss] = (_Float16)acc[nt][r];
      }
    }
  }
}

// ---------------------------------------------------------------------------
// Kernel 2: flash attention with double-buffered async K/V^T staging.
// grid = B * (S/64) = 256 blocks of 128 threads (4 waves).
// Wave w owns query rows [q0+16w, q0+16w+16) x full D=512.
// ---------------------------------------------------------------------------
__global__ __launch_bounds__(128) void flash_attn_kernel(
    const _Float16* __restrict__ q, const _Float16* __restrict__ k,
    const _Float16* __restrict__ vt, float* __restrict__ out) {
  __shared__ _Float16 Ks[2][32 * QK_STRIDE];    // 2 x (32 keys x 512 feats)
  __shared__ _Float16 Vts[2][DIM * V_STRIDE];   // 2 x (512 feats x 32 keys)
  __shared__ _Float16 Ps[4 * 16 * P_STRIDE];    // per-wave P scratch

  const int tid  = threadIdx.x;
  const int wave = tid >> 5;
  const int lane = tid & 31;
  const int hi   = lane >> 4;
  const int ln   = lane & 15;

  const int b  = blockIdx.x >> 6;          // 64 query blocks per batch
  const int q0 = (blockIdx.x & 63) * 64;   // first query row (within batch)
  const int qw = q0 + wave * 16;           // wave's first query row

  // Per-thread staging coordinates.
  const int kstage_row = tid >> 2;         // 0..31 (key row)
  const int kstage_col = (tid & 3) << 7;   // 0,128,256,384 (feature col)
  const int vstage_row = tid << 2;         // 4 feature rows per thread

  // Issue async copies for the K/V^T tile at key offset kb into buffer `buf`.
  auto stage_tile = [&](int kb, int buf) {
    const _Float16* ksrc =
        k + ((size_t)b * SEQ + kb + kstage_row) * DIM + kstage_col;
    _Float16* kdst = &Ks[buf][kstage_row * QK_STRIDE + kstage_col];
#pragma unroll
    for (int i = 0; i < 16; ++i) async_copy_b128(ksrc + i * 8, kdst + i * 8);
#pragma unroll
    for (int j = 0; j < 4; ++j) {
      const _Float16* vsrc =
          vt + ((size_t)b * DIM + vstage_row + j) * SEQ + kb;
      _Float16* vdst = &Vts[buf][(vstage_row + j) * V_STRIDE];
#pragma unroll
      for (int i = 0; i < 4; ++i) async_copy_b128(vsrc + i * 8, vdst + i * 8);
    }
  };

  // Q fragments held in registers: 16 chunks of K=32.
  const _Float16* qbase = q + ((size_t)b * SEQ + qw) * DIM;
  v16h qf[16];
#pragma unroll
  for (int c = 0; c < 16; ++c) qf[c] = load_frag_a_glb(qbase, DIM, c * 32, lane);

  v8f acc[32] = {};
  float mrow[8], lrow[8];
#pragma unroll
  for (int r = 0; r < 8; ++r) { mrow[r] = -INFINITY; lrow[r] = 0.0f; }

  _Float16* pw = Ps + wave * 16 * P_STRIDE;
  const int kend = q0 + 64;

  // Prologue: fill buffer 0.
  stage_tile(0, 0);
  async_wait0();
  __syncthreads();

  for (int kb = 0; kb < kend; kb += 32) {
    const int buf = (kb >> 5) & 1;
    // Kick off the next tile's copy while we compute on this one.
    if (kb + 32 < kend) stage_tile(kb + 32, buf ^ 1);

    if (kb <= qw + 15) {  // block has at least one unmasked key for this wave
      const _Float16* ks  = Ks[buf];
      const _Float16* vts = Vts[buf];

      // ---- scores: S(16x32) = Q(16x512) @ K^T(512x32), two 16x16 tiles ----
      v8f s0 = {}, s1 = {};
#pragma unroll
      for (int c = 0; c < 16; ++c) {
        v16h b0 = load_frag_b_lds(ks, QK_STRIDE, 0,  c * 32, lane);
        v16h b1 = load_frag_b_lds(ks, QK_STRIDE, 16, c * 32, lane);
        s0 = WMMA_F16(qf[c], b0, s0);
        s1 = WMMA_F16(qf[c], b1, s1);
      }

      // ---- causal mask (only needed when block straddles the diagonal) ----
      if (kb + 31 > qw) {
#pragma unroll
        for (int r = 0; r < 8; ++r) {
          const int qrow = qw + r + 8 * hi;
          if (kb + ln > qrow)      s0[r] = -INFINITY;
          if (kb + 16 + ln > qrow) s1[r] = -INFINITY;
        }
      }

      // ---- online softmax update ----
#pragma unroll
      for (int r = 0; r < 8; ++r) {
        float mx = fmaxf(s0[r], s1[r]);
        mx = fmaxf(mx, __shfl_xor(mx, 1, 32));
        mx = fmaxf(mx, __shfl_xor(mx, 2, 32));
        mx = fmaxf(mx, __shfl_xor(mx, 4, 32));
        mx = fmaxf(mx, __shfl_xor(mx, 8, 32));
        const float mn = fmaxf(mrow[r], mx);
        const float alpha = __expf(mrow[r] - mn);
        const float e0 = __expf(s0[r] - mn);
        const float e1 = __expf(s1[r] - mn);
        float ls = e0 + e1;
        ls += __shfl_xor(ls, 1, 32);
        ls += __shfl_xor(ls, 2, 32);
        ls += __shfl_xor(ls, 4, 32);
        ls += __shfl_xor(ls, 8, 32);
        lrow[r] = lrow[r] * alpha + ls;
        mrow[r] = mn;
#pragma unroll
        for (int nt = 0; nt < 32; ++nt) acc[nt][r] *= alpha;
        // stash P (f16) into per-wave LDS scratch for layout conversion
        pw[(r + 8 * hi) * P_STRIDE + ln]      = (_Float16)e0;
        pw[(r + 8 * hi) * P_STRIDE + 16 + ln] = (_Float16)e1;
      }
      asm volatile("" ::: "memory");  // same-wave LDS ops are in-order

      // ---- O += P(16x32) @ V(32x512): 32 output tiles ----
      v16h pa = load_frag_a_lds(pw, P_STRIDE, 0, 0, lane);
#pragma unroll
      for (int nt = 0; nt < 32; ++nt) {
        v16h bv = load_frag_b_lds(vts, V_STRIDE, nt * 16, 0, lane);
        acc[nt] = WMMA_F16(pa, bv, acc[nt]);
      }
    }

    async_wait0();      // next tile's copies have landed in LDS
    __syncthreads();    // everyone done reading cur / writing nxt
  }

  // ---- epilogue: O /= l, store f32 ----
  float* ob = out + ((size_t)b * SEQ + qw) * DIM;
#pragma unroll
  for (int r = 0; r < 8; ++r) {
    const float inv = 1.0f / lrow[r];
#pragma unroll
    for (int nt = 0; nt < 32; ++nt) {
      ob[(size_t)(r + 8 * hi) * DIM + nt * 16 + ln] = acc[nt][r] * inv;
    }
  }
}

// ---------------------------------------------------------------------------
extern "C" void kernel_launch(void* const* d_in, const int* in_sizes, int n_in,
                              void* d_out, int out_size, void* d_ws, size_t ws_size,
                              hipStream_t stream) {
  const float* x  = (const float*)d_in[0];
  const float* wq = (const float*)d_in[1];
  const float* wk = (const float*)d_in[2];
  const float* wv = (const float*)d_in[3];

  _Float16* qws  = (_Float16*)d_ws;                    // 16 MB
  _Float16* kws  = qws + (size_t)MTOT * DIM;           // 16 MB
  _Float16* vtws = kws + (size_t)MTOT * DIM;           // 16 MB (transposed)

  dim3 gproj(2048, 3);
  qkv_proj_kernel<<<gproj, 128, 0, stream>>>(x, wq, wk, wv, qws, kws, vtws);

  flash_attn_kernel<<<BATCH * (SEQ / 64), 128, 0, stream>>>(
      qws, kws, vtws, (float*)d_out);
}
